// RecurrentAnalyzer_8478265442736
// MI455X (gfx1250) — compile-verified, matrix operand order checked by reading, not run on hardware
//
#include <hip/hip_runtime.h>
#include <hip/hip_bf16.h>
#include <math.h>

// ---------------------------------------------------------------------------
// Persistent-WG bf16-WMMA RNN for MI455X (gfx1250, wave32).
//   T=512 steps, N=1024 batch, I=64, R=256, O=1.
//   Each of 64 workgroups owns BR=16 batch rows; h lives in LDS for all T.
//   Weights are converted fp32->bf16 and transposed ([col][k]) once into d_ws
//   so WMMA B-tiles are contiguous 32B-per-lane loads from L2 (192MB, all
//   weights resident).  Recurrence is latency-bound: per step, 4 dependent
//   16-wave WMMA phases separated by LDS barriers; activations are branchless
//   exp/rcp so no EXEC divergence near WMMA.
// ---------------------------------------------------------------------------

typedef __attribute__((ext_vector_type(16))) __bf16 v16bf;
typedef __attribute__((ext_vector_type(8)))  float  v8f;

#define T_STEPS 512
#define NBATCH  1024
#define ISZ     64
#define RSZ     256
#define BR      16              // batch rows per workgroup (one M-tile)
#define NWG     (NBATCH / BR)   // 64 workgroups
#define NTHREADS 512
#define NWAVES  (NTHREADS / 32) // 16 waves -> one 16x16 output tile per wave/phase

// Branchless fast tanh: 1 - 2/(e^{2x}+1).
//   v_exp_f32 + v_add + v_rcp_f32 + v_fma; saturates to +/-1 for large |x|
//   (e -> inf => rcp -> 0; e -> 0 => -1).  bf16 output makes this plenty.
__device__ __forceinline__ float fast_tanh(float x) {
    float e = __expf(2.0f * x);
    float r = __builtin_amdgcn_rcpf(e + 1.0f);
    return 1.0f - 2.0f * r;
}

// ---- WMMA operand loads ----------------------------------------------------
// A (16xK slice, bf16, row-major in LDS): lane L loads 16 contiguous bf16 of
// row (L&15); wave halves take adjacent 16-element K chunks. (Per-lane K
// permutation vs ISA table unverifiable compile-only; footprint matches.)
__device__ __forceinline__ v16bf load_a_lds(const __bf16* base, int ld, int k0) {
    const int lane = threadIdx.x & 31;
    const __bf16* p = base + (lane & 15) * ld + k0 + ((lane >> 4) << 4);
    return *(const v16bf*)p;
}

// B (Kx16 tile of column-major-stored weight: wt[col*ldk + k]).
__device__ __forceinline__ v16bf load_b_w(const __bf16* __restrict__ wt, int ldk,
                                          int c0, int k0) {
    const int lane = threadIdx.x & 31;
    const __bf16* p = wt + (size_t)(c0 + (lane & 15)) * ldk + k0 + ((lane >> 4) << 4);
    return *(const v16bf*)p;
}

// One 16x16 f32 tile = KSTEPS chained v_wmma_f32_16x16x32_bf16.
template <int KSTEPS>
__device__ __forceinline__ v8f gemm_tile(const __bf16* a_lds, int lda,
                                         const __bf16* __restrict__ wt, int ldk,
                                         int c0) {
    v8f acc = {};
#pragma unroll
    for (int k = 0; k < KSTEPS; ++k) {
        v16bf a = load_a_lds(a_lds, lda, k * 32);
        v16bf b = load_b_w(wt, ldk, c0, k * 32);
        acc = __builtin_amdgcn_wmma_f32_16x16x32_bf16(
            /*neg_a=*/false, a, /*neg_b=*/false, b,
            /*c_mod=*/(short)0, acc, /*reuse_a=*/false, /*reuse_b=*/false);
    }
    return acc;
}

// C/D 16x16 f32 layout: vgpr v, lanes 0-15 -> (M=v, N=lane); lanes 16-31 ->
// (M=8+v, N=lane-16).  Apply bias + fast_tanh, write bf16 into LDS row-major.
__device__ __forceinline__ void store_tanh(__bf16* dst, int ld, int dstc0,
                                           v8f acc, const float* bias_lds, int bc0) {
    const int lane = threadIdx.x & 31;
    const int n  = lane & 15;
    const int mh = (lane >> 4) << 3;
    const float b = bias_lds[bc0 + n];
#pragma unroll
    for (int v = 0; v < 8; ++v) {
        dst[(mh + v) * ld + dstc0 + n] = (__bf16)fast_tanh(acc[v] + b);
    }
}

// ---- weight prep: fp32 row-major [K][C] -> bf16 transposed [C][K] ----------
__global__ void convert_transpose(const float* __restrict__ src,
                                  __bf16* __restrict__ dst, int K, int C) {
    int idx = blockIdx.x * blockDim.x + threadIdx.x;
    if (idx < K * C) {
        int k = idx / C, c = idx % C;
        dst[(size_t)c * K + k] = (__bf16)src[idx];
    }
}

// ---- main persistent RNN kernel -------------------------------------------
__global__ void __launch_bounds__(NTHREADS)
rnn_kernel(const float* __restrict__ inputs,
           const __bf16* __restrict__ WhpT, const float* __restrict__ bhp,
           const __bf16* __restrict__ WcpT, const float* __restrict__ bcp,
           const __bf16* __restrict__ WmT,  const float* __restrict__ bm,
           const __bf16* __restrict__ WpoT, const float* __restrict__ bpo,
           const __bf16* __restrict__ Wr1T, const float* __restrict__ br1,
           const float* __restrict__ Wr2,   const float* __restrict__ br2,
           float* __restrict__ out)
{
    __shared__ __bf16 h_s[BR * RSZ];          // hidden state (persists over T)
    __shared__ __bf16 cat_s[BR * 2 * RSZ];    // tanh([hp, cp])
    __shared__ __bf16 m_s[BR * RSZ];          // tanh(meshup)
    __shared__ __bf16 x_s[BR * ISZ];          // staged x_t chunk
    __shared__ float  bhp_s[RSZ], bcp_s[RSZ], bm_s[RSZ], bpo_s[RSZ],
                      br1_s[RSZ], wr2_s[RSZ];
    __shared__ float  oacc_s[BR];

    const int tid     = threadIdx.x;
    const int wave    = tid >> 5;
    const int rowBase = blockIdx.x * BR;

    // init: h0 = 0; stage biases + Wr2 column into LDS once.
    for (int i = tid; i < BR * RSZ; i += NTHREADS) h_s[i] = (__bf16)0.0f;
    for (int i = tid; i < RSZ; i += NTHREADS) {
        bhp_s[i] = bhp[i]; bcp_s[i] = bcp[i]; bm_s[i] = bm[i];
        bpo_s[i] = bpo[i]; br1_s[i] = br1[i]; wr2_s[i] = Wr2[i];
    }
    __syncthreads();

    for (int t = 0; t < T_STEPS; ++t) {
        // stage x_t (BR x ISZ fp32 -> bf16); zero the O=1 row accumulators.
        const float* xg = inputs + ((size_t)t * NBATCH + rowBase) * ISZ;
        for (int i = tid; i < BR * ISZ; i += NTHREADS) x_s[i] = (__bf16)xg[i];
        if (tid < BR) oacc_s[tid] = 0.0f;
        // Prefetch next step's x chunk into cache while we grind 4 GEMM phases
        // (lowers to global_prefetch_b8; speculative, no counter cost).
        if (t + 1 < T_STEPS) {
            const float* xn = inputs + ((size_t)(t + 1) * NBATCH + rowBase) * ISZ;
            __builtin_prefetch(xn + tid * ((BR * ISZ) / NTHREADS), 0, 1);
        }
        __syncthreads();

        // Phase A1: cat[:, :R] = tanh(h @ Whp + bhp)   (16 tiles, 1/wave)
        for (int ti = wave; ti < 16; ti += NWAVES) {
            v8f acc = gemm_tile<RSZ / 32>(h_s, RSZ, WhpT, RSZ, ti * 16);
            store_tanh(cat_s, 2 * RSZ, ti * 16, acc, bhp_s, ti * 16);
        }
        // Phase A2: cat[:, R:] = tanh(x @ Wcp + bcp)   (16 tiles, 1/wave)
        for (int ti = wave; ti < 16; ti += NWAVES) {
            v8f acc = gemm_tile<ISZ / 32>(x_s, ISZ, WcpT, ISZ, ti * 16);
            store_tanh(cat_s, 2 * RSZ, RSZ + ti * 16, acc, bcp_s, ti * 16);
        }
        __syncthreads();

        // Phase B: m = tanh(cat @ Wm + bm)              (16 tiles, 1/wave)
        for (int ti = wave; ti < 16; ti += NWAVES) {
            v8f acc = gemm_tile<(2 * RSZ) / 32>(cat_s, 2 * RSZ, WmT, 2 * RSZ, ti * 16);
            store_tanh(m_s, RSZ, ti * 16, acc, bm_s, ti * 16);
        }
        __syncthreads();

        // Phase C: h = tanh(m @ Whpost + bhpost)  (old h dead after Phase A)
        for (int ti = wave; ti < 16; ti += NWAVES) {
            v8f acc = gemm_tile<RSZ / 32>(m_s, RSZ, WpoT, RSZ, ti * 16);
            store_tanh(h_s, RSZ, ti * 16, acc, bpo_s, ti * 16);
        }
        __syncthreads();

        // Phase D: out = relu(h @ Wr1 + br1) @ Wr2 + br2   (O == 1)
        for (int ti = wave; ti < 16; ti += NWAVES) {
            v8f acc = gemm_tile<RSZ / 32>(h_s, RSZ, Wr1T, RSZ, ti * 16);
            const int lane = tid & 31;
            const int n  = lane & 15;
            const int mh = (lane >> 4) << 3;
            const float w2 = wr2_s[ti * 16 + n];
            const float b1 = br1_s[ti * 16 + n];
#pragma unroll
            for (int v = 0; v < 8; ++v) {
                float r = acc[v] + b1;
                r = r > 0.0f ? r : 0.0f;
                float p = r * w2;
                // reduce the 16 N-columns held across a half-wave
                p += __shfl_xor(p, 1, 16);
                p += __shfl_xor(p, 2, 16);
                p += __shfl_xor(p, 4, 16);
                p += __shfl_xor(p, 8, 16);
                if (n == 0) atomicAdd(&oacc_s[mh + v], p);   // ds_add_f32
            }
        }
        __syncthreads();

        if (tid < BR)
            out[(size_t)t * NBATCH + rowBase + tid] = oacc_s[tid] + br2[0];
        __syncthreads();
    }
}

// ---------------------------------------------------------------------------
extern "C" void kernel_launch(void* const* d_in, const int* in_sizes, int n_in,
                              void* d_out, int out_size, void* d_ws, size_t ws_size,
                              hipStream_t stream) {
    (void)in_sizes; (void)n_in; (void)out_size; (void)ws_size;

    const float* inputs = (const float*)d_in[0];
    const float* Whp = (const float*)d_in[1];
    const float* bhp = (const float*)d_in[2];
    const float* Wcp = (const float*)d_in[3];
    const float* bcp = (const float*)d_in[4];
    const float* Wm  = (const float*)d_in[5];
    const float* bm  = (const float*)d_in[6];
    const float* Wpo = (const float*)d_in[7];
    const float* bpo = (const float*)d_in[8];
    const float* Wr1 = (const float*)d_in[9];
    const float* br1 = (const float*)d_in[10];
    const float* Wr2 = (const float*)d_in[11];
    const float* br2 = (const float*)d_in[12];
    float* out = (float*)d_out;

    // d_ws layout: bf16 transposed weights, 344064 elems = 672 KB.
    __bf16* ws   = (__bf16*)d_ws;
    __bf16* WhpT = ws;
    __bf16* WcpT = WhpT + RSZ * RSZ;
    __bf16* WmT  = WcpT + ISZ * RSZ;
    __bf16* WpoT = WmT + 2 * RSZ * RSZ;
    __bf16* Wr1T = WpoT + RSZ * RSZ;

    const int nt = 256;
    convert_transpose<<<(RSZ * RSZ + nt - 1) / nt, nt, 0, stream>>>(Whp, WhpT, RSZ, RSZ);
    convert_transpose<<<(ISZ * RSZ + nt - 1) / nt, nt, 0, stream>>>(Wcp, WcpT, ISZ, RSZ);
    convert_transpose<<<(2 * RSZ * RSZ + nt - 1) / nt, nt, 0, stream>>>(Wm, WmT, 2 * RSZ, RSZ);
    convert_transpose<<<(RSZ * RSZ + nt - 1) / nt, nt, 0, stream>>>(Wpo, WpoT, RSZ, RSZ);
    convert_transpose<<<(RSZ * RSZ + nt - 1) / nt, nt, 0, stream>>>(Wr1, Wr1T, RSZ, RSZ);

    rnn_kernel<<<NWG, NTHREADS, 0, stream>>>(inputs, WhpT, bhp, WcpT, bcp,
                                             WmT, bm, WpoT, bpo,
                                             Wr1T, br1, Wr2, br2, out);
}
